// FlashAttnGQADecodeKVCache_41592463294629
// MI455X (gfx1250) — matrix operand order repaired and checked by reading, not compiled, not run on hardware
//
#include <hip/hip_runtime.h>
#include <hip/hip_bf16.h>
#include <stdint.h>

// -----------------------------------------------------------------------------
// GQA flash-decode for MI455X (gfx1250), wave32 + WMMA bf16.
// B=16, H=32, D=128, G=8 (KVG=4 queries per kv head), S=8192.
// Memory-bound (~0.6-1.1 GB KV traffic vs ~2.2 GFLOP): bf16 WMMA with fp32
// accumulate; split-K flash-decode (up to 32 splits -> 4096 waves) with LSE
// combine. K/V are converted f32->bf16 ONCE at tile-load (packed
// v_cvt_pk_bf16_f32) and staged in LDS as bf16, so WMMA fragment builds are
// pure ds_load_b128 / b16 with no VALU.
// -----------------------------------------------------------------------------

namespace {
constexpr int kB = 16, kH = 32, kD = 128, kG = 8, kKVG = 4, kS = 8192;
constexpr int kBG = kB * kG;
constexpr float kScale = 0.08838834764831845f;  // 1/sqrt(128)
}

typedef __attribute__((ext_vector_type(16))) __bf16 v16bf;
typedef __attribute__((ext_vector_type(8)))  __bf16 v8bf;
typedef __attribute__((ext_vector_type(2)))  __bf16 v2bf;
typedef __attribute__((ext_vector_type(8)))  float  v8f;

// Two f32 -> packed bf16 pair (native casts; lowers to v_cvt_pk_bf16_f32).
__device__ __forceinline__ uint32_t pack2(float a, float b) {
  v2bf t; t[0] = (__bf16)a; t[1] = (__bf16)b;
  return __builtin_bit_cast(uint32_t, t);
}

// One wave (32 threads) per workgroup. grid = (B*G, nsplit).
__global__ __launch_bounds__(32)
void fa_gqa_decode_kernel(const float* __restrict__ Qp, const float* __restrict__ Kp,
                          const float* __restrict__ Vp, const int* __restrict__ seqlens,
                          float* __restrict__ Opart, float* __restrict__ Mpart,
                          float* __restrict__ Lpart, float* __restrict__ Out,
                          int nsplit, int direct) {
  __shared__ uint32_t LK[32 * 64];   // K tile [key32][dpair64] bf16x2 (8 KB)
  __shared__ uint32_t LV[32 * 64];   // V tile [key32][dpair64] bf16x2 (8 KB)
  __shared__ uint32_t LP[16 * 16];   // probs  [q16][keypair16] bf16x2 (1 KB)

  const int bg    = blockIdx.x;
  const int b     = bg >> 3;         // / kG
  const int g     = bg & 7;
  const int split = blockIdx.y;
  const int lane  = threadIdx.x;
  const int nq    = lane & 15;       // N column (q slot) / key-within-subtile
  const int h16   = lane >> 4;       // lane half

  const int seqlen = seqlens[b];
  const int chunk  = (seqlen + nsplit - 1) / nsplit;
  const int kbeg   = split * chunk;
  const int kend   = min(kbeg + chunk, seqlen);

  if (kbeg >= kend) {                // empty split: neutral partials
    if (!direct) {
      const size_t sb = ((size_t)split * kBG + bg) * kKVG;
      if (lane < 16)
        for (int r = 0; r < 4; ++r)
          for (int t = 0; t < 8; ++t)
            Opart[(sb + r) * kD + 16 * t + lane] = 0.f;
      if (lane < 4) { Mpart[sb + lane] = -1e30f; Lpart[sb + lane] = 0.f; }
    }
    return;
  }

  // ---- Q as WMMA B-operand (32x16 bf16): lane n holds column q=n&15,
  //      K-rows d = 32c + 16*h16 + (0..15) contiguous, same order as the K A-operand. ----
  v16bf qf[4];
  {
    const bool qv = (nq < kKVG);
    const int  hq = g * kKVG + (qv ? nq : 0);
    const float* qrow = Qp + ((size_t)b * kH + hq) * kD;
#pragma unroll
    for (int c = 0; c < 4; ++c) {
      const int d0 = 32 * c + 16 * h16;
      v16bf v;
      if (qv) {
        const float4 f0 = *(const float4*)(qrow + d0);
        const float4 f1 = *(const float4*)(qrow + d0 + 4);
        const float4 f2 = *(const float4*)(qrow + d0 + 8);
        const float4 f3 = *(const float4*)(qrow + d0 + 12);
        v[0] = (__bf16)f0.x;  v[1] = (__bf16)f0.y;  v[2] = (__bf16)f0.z;  v[3] = (__bf16)f0.w;
        v[4] = (__bf16)f1.x;  v[5] = (__bf16)f1.y;  v[6] = (__bf16)f1.z;  v[7] = (__bf16)f1.w;
        v[8] = (__bf16)f2.x;  v[9] = (__bf16)f2.y;  v[10] = (__bf16)f2.z; v[11] = (__bf16)f2.w;
        v[12] = (__bf16)f3.x; v[13] = (__bf16)f3.y; v[14] = (__bf16)f3.z; v[15] = (__bf16)f3.w;
      } else {
#pragma unroll
        for (int e = 0; e < 16; ++e) v[e] = (__bf16)0.f;
      }
      qf[c] = v;
    }
  }

  v8f acc[8];                        // out [q=M rows][d=N], 8 d-tiles of 16
#pragma unroll
  for (int t = 0; t < 8; ++t)
#pragma unroll
    for (int r = 0; r < 8; ++r) acc[t][r] = 0.f;

  float m_run = -1e30f, l_run = 0.f; // per-q (q = lane&15) online-softmax state

  for (int k0 = kbeg; k0 < kend; k0 += 32) {
    const int nvalid = min(32, kend - k0);
    __syncthreads();                 // LDS reuse fence

    // ---- coalesced tile load: row i, 32 lanes cover 32 float4 columns;
    //      convert packed f32->bf16 once, store bf16 pairs to LDS ----
#pragma unroll 4
    for (int i = 0; i < 32; ++i) {
      int kr = k0 + i; if (kr >= seqlen) kr = seqlen - 1;
      const size_t ro = (((size_t)b * kS + kr) * kG + g) * (size_t)kD + lane * 4;
      const float4 kk = *(const float4*)(Kp + ro);
      const float4 vv = *(const float4*)(Vp + ro);
      *(uint2*)&LK[i * 64 + lane * 2] = make_uint2(pack2(kk.x, kk.y), pack2(kk.z, kk.w));
      *(uint2*)&LV[i * 64 + lane * 2] = make_uint2(pack2(vv.x, vv.y), pack2(vv.z, vv.w));
      if (k0 + 32 < kend) {          // prefetch next tile (global_prefetch_b8)
        int k2 = min(k0 + 32 + i, seqlen - 1);
        const size_t r2 = (((size_t)b * kS + k2) * kG + g) * (size_t)kD + lane * 4;
        __builtin_prefetch(Kp + r2, 0, 0);
        __builtin_prefetch(Vp + r2, 0, 0);
      }
    }
    __syncthreads();

    // ---- scores^T[key=M, q=N] = K_tile x Q^T, two 16-key subtiles.
    //      A fragment = two ds_load_b128 of bf16 pairs, zero VALU. ----
    v8f st[2];
#pragma unroll
    for (int sub = 0; sub < 2; ++sub) {
      v8f sacc;
#pragma unroll
      for (int r = 0; r < 8; ++r) sacc[r] = 0.f;
      const uint32_t* krow = &LK[(sub * 16 + nq) * 64];
#pragma unroll
      for (int c = 0; c < 4; ++c) {
        // elems 0-15 -> d = 32c+16*h16+e (pair idx 16c+8*h16)
        const int p0 = 16 * c + 8 * h16;
        const uint4 lo = *(const uint4*)(krow + p0);
        const uint4 hi = *(const uint4*)(krow + p0 + 4);
        const v16bf af = __builtin_shufflevector(
            __builtin_bit_cast(v8bf, lo), __builtin_bit_cast(v8bf, hi),
            0, 1, 2, 3, 4, 5, 6, 7, 8, 9, 10, 11, 12, 13, 14, 15);
        sacc = __builtin_amdgcn_wmma_f32_16x16x32_bf16(false, af, false, qf[c],
                                                       (short)0, sacc, false, false);
      }
      st[sub] = sacc;
    }

    // ---- online softmax: column q = lane&15; row key = r + 8*h16 + 16*sub ----
    float tmax = -1e30f;
#pragma unroll
    for (int sub = 0; sub < 2; ++sub)
#pragma unroll
      for (int r = 0; r < 8; ++r) {
        const int kl = sub * 16 + 8 * h16 + r;
        float s = (kl < nvalid) ? st[sub][r] * kScale : -1e30f;
        st[sub][r] = s;
        tmax = fmaxf(tmax, s);
      }
    tmax = fmaxf(tmax, __shfl_xor(tmax, 16));
    const float mnew = fmaxf(m_run, tmax);
    const float corr = __expf(m_run - mnew);
    float lsum = 0.f;
#pragma unroll
    for (int sub = 0; sub < 2; ++sub)
#pragma unroll
      for (int rp = 0; rp < 4; ++rp) {
        const float p0 = __expf(st[sub][2 * rp]     - mnew);
        const float p1 = __expf(st[sub][2 * rp + 1] - mnew);
        lsum += p0 + p1;
        // key = sub*16 + 8*h16 + 2rp  ->  pair index sub*8 + 4*h16 + rp
        LP[nq * 16 + sub * 8 + 4 * h16 + rp] = pack2(p0, p1);
      }
    lsum += __shfl_xor(lsum, 16);
    l_run = l_run * corr + lsum;
    m_run = mnew;

    // ---- rescale accumulator rows (row q = r + 8*h16; stats live in lane q) --
#pragma unroll
    for (int r = 0; r < 8; ++r) {
      const float f = __shfl(corr, r + 8 * h16);
#pragma unroll
      for (int t = 0; t < 8; ++t) acc[t][r] *= f;
    }

    __syncthreads();
    // ---- probs as A operand 16(q) x 32(key): elems 0-15 -> key 16*h16+e ----
    const uint4 pa0 = *(const uint4*)&LP[nq * 16 + 8 * h16];
    const uint4 pa1 = *(const uint4*)&LP[nq * 16 + 8 * h16 + 4];
    const v16bf pf  = __builtin_shufflevector(
        __builtin_bit_cast(v8bf, pa0), __builtin_bit_cast(v8bf, pa1),
        0, 1, 2, 3, 4, 5, 6, 7, 8, 9, 10, 11, 12, 13, 14, 15);

    // ---- PV: B = V 32(key) x 16(d); lane col d = 16t + nq, elem e -> key 16*h16+e.
    //      Direct bf16 (b16) LDS reads, zero converts. ----
    const __bf16* LVh = (const __bf16*)LV;
#pragma unroll
    for (int t = 0; t < 8; ++t) {
      const int dcol = 16 * t + nq;
      v16bf vf;
#pragma unroll
      for (int e = 0; e < 16; ++e)
        vf[e] = LVh[(16 * h16 + e) * kD + dcol];
      acc[t] = __builtin_amdgcn_wmma_f32_16x16x32_bf16(false, pf, false, vf,
                                                       (short)0, acc[t], false, false);
    }
  }

  // ---- epilogue: valid q rows are r=0..3 in lanes 0..15 (h16==0) ----
  if (direct) {
    float inv[4];
    const float il = 1.f / l_run;
#pragma unroll
    for (int r = 0; r < 4; ++r) inv[r] = __shfl(il, r);
    if (lane < 16) {
#pragma unroll
      for (int r = 0; r < 4; ++r)
#pragma unroll
        for (int t = 0; t < 8; ++t)
          Out[((size_t)b * kH + g * kKVG + r) * kD + 16 * t + lane] = acc[t][r] * inv[r];
    }
  } else {
    const size_t sb = ((size_t)split * kBG + bg) * kKVG;
    if (lane < 16) {
#pragma unroll
      for (int r = 0; r < 4; ++r)
#pragma unroll
        for (int t = 0; t < 8; ++t)
          Opart[(sb + r) * kD + 16 * t + lane] = acc[t][r];
    }
    if (lane < 4) { Mpart[sb + lane] = m_run; Lpart[sb + lane] = l_run; }
  }
}

// LSE combine over splits. grid = B*H blocks, 128 threads (one per d).
__global__ __launch_bounds__(128)
void fa_gqa_combine_kernel(const float* __restrict__ Opart, const float* __restrict__ Mpart,
                           const float* __restrict__ Lpart, float* __restrict__ Out,
                           int nsplit) {
  const int bh = blockIdx.x;          // 0..511
  const int d  = threadIdx.x;         // 0..127
  const int b  = bh / kH, h = bh % kH;
  const int g  = h / kKVG, q = h % kKVG;
  const int bg = b * kG + g;

  float M = -1e30f;
  for (int s = 0; s < nsplit; ++s)
    M = fmaxf(M, Mpart[((size_t)s * kBG + bg) * kKVG + q]);
  float L = 0.f, o = 0.f;
  for (int s = 0; s < nsplit; ++s) {
    const size_t sq = ((size_t)s * kBG + bg) * kKVG + q;
    const float w = __expf(Mpart[sq] - M);
    L += Lpart[sq] * w;
    o += Opart[sq * kD + d] * w;
  }
  Out[(size_t)bh * kD + d] = o / L;
}

extern "C" void kernel_launch(void* const* d_in, const int* in_sizes, int n_in,
                              void* d_out, int out_size, void* d_ws, size_t ws_size,
                              hipStream_t stream) {
  const float* Qp = (const float*)d_in[0];
  const float* Kp = (const float*)d_in[1];
  const float* Vp = (const float*)d_in[2];
  const int*   sl = (const int*)d_in[3];
  float*       Out = (float*)d_out;

  const size_t perSplit =
      ((size_t)kBG * kKVG * kD + 2 * (size_t)kBG * kKVG) * sizeof(float);
  int nsplit = 32;                   // 4096 one-wave WGs -> saturate 23.3 TB/s
  while (nsplit > 1 && (size_t)nsplit * perSplit > ws_size) nsplit >>= 1;
  if ((size_t)nsplit * perSplit > ws_size) nsplit = 1;

  if (nsplit == 1) {
    fa_gqa_decode_kernel<<<dim3(kBG, 1), 32, 0, stream>>>(
        Qp, Kp, Vp, sl, nullptr, nullptr, nullptr, Out, 1, 1);
  } else {
    float* Opart = (float*)d_ws;
    float* Mpart = Opart + (size_t)nsplit * kBG * kKVG * kD;
    float* Lpart = Mpart + (size_t)nsplit * kBG * kKVG;
    fa_gqa_decode_kernel<<<dim3(kBG, nsplit), 32, 0, stream>>>(
        Qp, Kp, Vp, sl, Opart, Mpart, Lpart, nullptr, nsplit, 0);
    fa_gqa_combine_kernel<<<dim3(kB * kH), 128, 0, stream>>>(
        Opart, Mpart, Lpart, Out, nsplit);
  }
}
